// FactorizedDenseAttention_45337674777144
// MI455X (gfx1250) — matrix operand order repaired and checked by reading, not compiled, not run on hardware
//
#include <hip/hip_runtime.h>
#include <hip/hip_bf16.h>
#include <stdint.h>

// ---------------------------------------------------------------------------
// FactorizedDenseAttention for MI455X (gfx1250, wave32, WMMA).
//
//   S = Atil @ Btil^T   (rank-32),  P = softmax_rows(S),  O = P @ V
//   outputs: (O [16,2048,64] f32, P [16,2048,2048] f32) concatenated.
//
// Roofline: ~13 GFLOP vs a 268 MB mandatory dense_attn write -> memory bound.
// Design: single pass over dense_attn (write-only, non-temporal), S recomputed
// per sweep (1 WMMA per 16x16 tile, K=32), factors/V held in L2.
// ---------------------------------------------------------------------------

typedef __attribute__((ext_vector_type(16))) _Float16 v16h;
typedef __attribute__((ext_vector_type(8)))  float    v8f;

#define H_    16
#define L_    2048
#define DK_   64

__device__ __forceinline__ v8f wmma_f16(v16h a, v16h b, v8f c) {
  return __builtin_amdgcn_wmma_f32_16x16x32_f16(
      /*neg_a=*/false, a, /*neg_b=*/false, b,
      /*c_mod=*/(short)0, c, /*reuse_a=*/false, /*reuse_b=*/false);
}

// A fragment (16x32 f16, M = lane%16): elems 0..15 = K (hi?16..31:0..15).
// Row is 32 halfs (64B).
__device__ __forceinline__ v16h load_fragA(const _Float16* row, int hi) {
  union { v16h v; float4 f[2]; } u;
  const char* p = (const char*)row;
  u.f[0] = *(const float4*)(p + hi * 32);
  u.f[1] = *(const float4*)(p + 16 + hi * 32);
  return u.v;
}

// B fragment (32x16 f16, N = lane%16): elems 0..15 = K (hi?16..31:0..15).
// "row" points at the N-th row of Btil (32 halfs): B[k,n] = Btil[n][k].
__device__ __forceinline__ v16h load_fragB(const _Float16* row, int hi) {
  union { v16h v; float4 f[2]; } u;
  const char* p = (const char*)row + hi * 32;
  u.f[0] = *(const float4*)(p);
  u.f[1] = *(const float4*)(p + 16);
  return u.v;
}

// ---------------------------------------------------------------------------
// Kernel 1: Atil[h,i,m] = f16( 32*(a[2m]+a[2m+1]) ),  Btil[h,i,m] = f16(b[m])
// one block (64 threads) per (h,i) row.
// ---------------------------------------------------------------------------
__global__ void fda_prep(const float* __restrict__ q,
                         const float* __restrict__ Wa, const float* __restrict__ ba,
                         const float* __restrict__ Wb, const float* __restrict__ bb,
                         _Float16* __restrict__ Atil, _Float16* __restrict__ Btil) {
  const int row = blockIdx.x;       // h*L + i
  const int t   = threadIdx.x;      // 0..63
  __shared__ float qrow[64];
  __shared__ float aout[64];
  qrow[t] = q[(size_t)row * 64 + t];
  __syncthreads();
  float acc = ba[t];
  #pragma unroll 8
  for (int k = 0; k < 64; ++k) acc += Wa[t * 64 + k] * qrow[k];
  aout[t] = acc;
  float bacc = 0.0f;
  if (t < 32) {
    bacc = bb[t];
    #pragma unroll 8
    for (int k = 0; k < 64; ++k) bacc += Wb[t * 64 + k] * qrow[k];
  }
  __syncthreads();
  if (t < 32) {
    Atil[(size_t)row * 32 + t] = (_Float16)(32.0f * (aout[2 * t] + aout[2 * t + 1]));
    Btil[(size_t)row * 32 + t] = (_Float16)bacc;
  }
}

// ---------------------------------------------------------------------------
// Kernel 2: pack V into f16 j-pairs:  Vp[h][d][jp] = pack(v[h,2jp,d], v[h,2jp+1,d])
// ---------------------------------------------------------------------------
__global__ void fda_packv(const float* __restrict__ v, uint32_t* __restrict__ Vp) {
  const int idx = blockIdx.x * 256 + threadIdx.x;   // (h*64+d)*1024 + jp
  const int jp = idx & 1023;
  const int hd = idx >> 10;
  const int d  = hd & 63;
  const int h  = hd >> 6;
  const float v0 = v[((size_t)h * L_ + 2 * jp) * DK_ + d];
  const float v1 = v[((size_t)h * L_ + 2 * jp + 1) * DK_ + d];
  union { uint32_t u; _Float16 h2[2]; } pk;
  pk.h2[0] = (_Float16)v0;
  pk.h2[1] = (_Float16)v1;
  Vp[idx] = pk.u;
}

// ---------------------------------------------------------------------------
// Kernel 3: fused  S -> softmax -> (write P, O = P@V).
// grid (128 row-blocks, 16 heads), 256 threads = 8 waves.
// Wave w owns column tiles [w*16, w*16+16).
// ---------------------------------------------------------------------------
__global__ __launch_bounds__(256)
void fda_attn(const _Float16* __restrict__ Atil, const _Float16* __restrict__ Btil,
              const uint32_t* __restrict__ Vp,
              float* __restrict__ outO, float* __restrict__ outP) {
  const int ib  = blockIdx.x;        // row block 0..127
  const int h   = blockIdx.y;        // head
  const int tid = threadIdx.x;
  const int w   = tid >> 5;          // wave 0..7
  const int l   = tid & 31;          // lane
  const int lh  = l & 15;            // lane % 16
  const int hi  = l >> 4;            // lane half

  __shared__ __align__(16) float    ldsO[8][16 * 64];   // per-wave O partials
  __shared__ __align__(16) _Float16 ldsP[8][16 * 32];   // per-wave P staging
  __shared__ float wstat[8][16];
  __shared__ float mrow[16];
  __shared__ float srow[16];

  const _Float16* btilH = Btil + (size_t)h * L_ * 32;
  const _Float16* arowp = Atil + ((size_t)h * L_ + ib * 16 + lh) * 32;
  const v16h afrag = load_fragA(arowp, hi);

  // ---- sweep 1: row max --------------------------------------------------
  float mmax[8];
  #pragma unroll
  for (int r = 0; r < 8; ++r) mmax[r] = -3.0e38f;
  for (int t = 0; t < 16; ++t) {
    const int jt = w * 16 + t;
    const v16h bfrag = load_fragB(btilH + (size_t)(jt * 16 + lh) * 32, hi);
    v8f c = {};
    c = wmma_f16(afrag, bfrag, c);
    #pragma unroll
    for (int r = 0; r < 8; ++r) mmax[r] = fmaxf(mmax[r], c[r]);
  }
  #pragma unroll
  for (int d = 1; d < 16; d <<= 1) {
    #pragma unroll
    for (int r = 0; r < 8; ++r) mmax[r] = fmaxf(mmax[r], __shfl_xor(mmax[r], d, 32));
  }
  if (lh == 0) {
    #pragma unroll
    for (int r = 0; r < 8; ++r) wstat[w][hi * 8 + r] = mmax[r];
  }
  __syncthreads();
  if (tid < 16) {
    float m = wstat[0][tid];
    #pragma unroll
    for (int ww = 1; ww < 8; ++ww) m = fmaxf(m, wstat[ww][tid]);
    mrow[tid] = m;
  }
  __syncthreads();
  float mr[8];
  #pragma unroll
  for (int r = 0; r < 8; ++r) mr[r] = mrow[hi * 8 + r];

  // ---- sweep 2: row sum of exp ------------------------------------------
  float ssum[8];
  #pragma unroll
  for (int r = 0; r < 8; ++r) ssum[r] = 0.0f;
  for (int t = 0; t < 16; ++t) {
    const int jt = w * 16 + t;
    const v16h bfrag = load_fragB(btilH + (size_t)(jt * 16 + lh) * 32, hi);
    v8f c = {};
    c = wmma_f16(afrag, bfrag, c);
    #pragma unroll
    for (int r = 0; r < 8; ++r) ssum[r] += __expf(c[r] - mr[r]);
  }
  #pragma unroll
  for (int d = 1; d < 16; d <<= 1) {
    #pragma unroll
    for (int r = 0; r < 8; ++r) ssum[r] += __shfl_xor(ssum[r], d, 32);
  }
  if (lh == 0) {
    #pragma unroll
    for (int r = 0; r < 8; ++r) wstat[w][hi * 8 + r] = ssum[r];
  }
  __syncthreads();
  if (tid < 16) {
    float s = 0.0f;
    #pragma unroll
    for (int ww = 0; ww < 8; ++ww) s += wstat[ww][tid];
    srow[tid] = s;
  }
  __syncthreads();
  float sinv[8];
  #pragma unroll
  for (int r = 0; r < 8; ++r) sinv[r] = 1.0f / srow[hi * 8 + r];

  // ---- sweep 3: write P (non-temporal), accumulate O = P @ V -------------
  v8f oacc[4];
  #pragma unroll
  for (int nb = 0; nb < 4; ++nb) oacc[nb] = (v8f){};

  _Float16* myP = ldsP[w];
  const size_t pbase = ((size_t)h * L_ + (size_t)ib * 16) * L_;
  const uint32_t* vpH = Vp + (size_t)h * 64 * 1024;

  for (int tp = 0; tp < 8; ++tp) {
    const int jt1 = w * 16 + 2 * tp;    // two adjacent 16-col tiles -> K=32
    #pragma unroll
    for (int half = 0; half < 2; ++half) {
      const int jt = jt1 + half;
      const v16h bfrag = load_fragB(btilH + (size_t)(jt * 16 + lh) * 32, hi);
      v8f c = {};
      c = wmma_f16(afrag, bfrag, c);
      #pragma unroll
      for (int r = 0; r < 8; ++r) {
        const float p = __expf(c[r] - mr[r]) * sinv[r];
        const int row = hi * 8 + r;
        // dense_attn: 268 MB streamed once -> non-temporal, keep L2 for Atil/Btil/Vp
        __builtin_nontemporal_store(p, &outP[pbase + (size_t)row * L_ + jt * 16 + lh]);
        myP[row * 32 + half * 16 + lh] = (_Float16)p;            // stage for PV
      }
    }
    __syncthreads();                       // LDS stores -> fragment loads
    const v16h pfrag = load_fragA(myP + lh * 32, hi);
    const int jp0 = (jt1 + hi) * 8;        // hi=0 -> jt1 pairs, hi=1 -> jt2 pairs
    #pragma unroll
    for (int nb = 0; nb < 4; ++nb) {
      const int d = nb * 16 + lh;
      const uint32_t* vp = vpH + (size_t)d * 1024 + jp0;
      union { v16h v; uint32_t u32[8]; } uv;
      #pragma unroll
      for (int r = 0; r < 8; ++r) uv.u32[r] = vp[r];
      oacc[nb] = wmma_f16(pfrag, uv.v, oacc[nb]);
    }
    __syncthreads();                       // guard LDS reuse next iteration
  }

  // ---- reduce O across waves and store -----------------------------------
  #pragma unroll
  for (int nb = 0; nb < 4; ++nb) {
    #pragma unroll
    for (int r = 0; r < 8; ++r)
      ldsO[w][(hi * 8 + r) * 64 + nb * 16 + lh] = oacc[nb][r];
  }
  __syncthreads();
  for (int e = tid; e < 16 * 64; e += 256) {
    float s = 0.0f;
    #pragma unroll
    for (int ww = 0; ww < 8; ++ww) s += ldsO[ww][e];
    const int row = e >> 6;
    const int d   = e & 63;
    __builtin_nontemporal_store(
        s, &outO[((size_t)h * L_ + (size_t)ib * 16 + row) * DK_ + d]);
  }
}

// ---------------------------------------------------------------------------
extern "C" void kernel_launch(void* const* d_in, const int* in_sizes, int n_in,
                              void* d_out, int out_size, void* d_ws, size_t ws_size,
                              hipStream_t stream) {
  const float* q  = (const float*)d_in[0];
  const float* v  = (const float*)d_in[1];
  const float* Wa = (const float*)d_in[2];
  const float* ba = (const float*)d_in[3];
  const float* Wb = (const float*)d_in[4];
  const float* bb = (const float*)d_in[5];
  // d_in[6] = len_q (== 2048, hardcoded)

  float* outO = (float*)d_out;                               // [16,2048,64]
  float* outP = outO + (size_t)H_ * L_ * DK_;                // [16,2048,2048]

  char* ws = (char*)d_ws;
  _Float16* Atil = (_Float16*)(ws);                          // 2 MB
  _Float16* Btil = (_Float16*)(ws + ((size_t)2 << 20));      // 2 MB
  uint32_t* Vp   = (uint32_t*)(ws + ((size_t)4 << 20));      // 4 MB

  fda_prep <<<dim3(H_ * L_), dim3(64), 0, stream>>>(q, Wa, ba, Wb, bb, Atil, Btil);
  fda_packv<<<dim3((H_ * 64 * 1024) / 256), dim3(256), 0, stream>>>(v, Vp);
  fda_attn <<<dim3(L_ / 16, H_), dim3(256), 0, stream>>>(Atil, Btil, Vp, outO, outP);
}
